// MultiHeadedAttention_9715216024224
// MI455X (gfx1250) — matrix operand order repaired and checked by reading, not compiled, not run on hardware
//
#include <hip/hip_runtime.h>
#include <hip/hip_bf16.h>

#define DEV __device__ __forceinline__

typedef __attribute__((ext_vector_type(16))) _Float16 v16h;
typedef __attribute__((ext_vector_type(8)))  _Float16 h8v;
typedef __attribute__((ext_vector_type(8)))  float    v8f;
typedef __attribute__((ext_vector_type(4)))  float    f4v;

static constexpr int BATCH  = 4;
static constexpr int HEADS  = 16;
static constexpr int SEQ    = 2048;
static constexpr int DK     = 64;
static constexpr int DMODEL = 1024;
static constexpr int MROWS  = BATCH * SEQ;   // 8192

// ---------------- fragment loaders (contiguous k = 16g+s, same order in A and B) ----------

DEV h8v cvt_h8(const float* p) {            // 8 f32 -> 8 f16 (32B-aligned source)
  f4v a = *(const f4v*)p;
  f4v b = *(const f4v*)(p + 4);
  h8v r;
#pragma unroll
  for (int i = 0; i < 4; ++i) { r[i] = (_Float16)a[i]; r[4 + i] = (_Float16)b[i]; }
  return r;
}

DEV v16h combine16(h8v a, h8v b) {
  v16h r;
#pragma unroll
  for (int i = 0; i < 8; ++i) { r[i] = a[i]; r[8 + i] = b[i]; }
  return r;
}

// A 16x32: lane(16g+m) = row m; elems = K[k0+16g .. k0+16g+16) contiguous
DEV v16h a_frag(const float* row, int k0, int g) {
  return combine16(cvt_h8(row + k0 + 16 * g), cvt_h8(row + k0 + 16 * g + 8));
}
DEV v16h a_frag(const _Float16* row, int k0, int g) {
  return combine16(*(const h8v*)(row + k0 + 16 * g),
                   *(const h8v*)(row + k0 + 16 * g + 8));
}
// B 32x16: lane(16g+n) = col n; elems = K[k16 .. k16+16) contiguous
DEV v16h b_frag(const float* col, int k16) {
  return combine16(cvt_h8(col + k16), cvt_h8(col + k16 + 8));
}
DEV v16h b_frag(const _Float16* col, int k16) {
  return combine16(*(const h8v*)(col + k16), *(const h8v*)(col + k16 + 8));
}

DEV v8f wmma_f16(v16h a, v16h b, v8f c) {
  return __builtin_amdgcn_wmma_f32_16x16x32_f16(false, a, false, b, (short)0, c,
                                                false, false);
}

// ---------------- GEMM: out = A @ W^T + bias --------------------------------
// A: [MROWS, DMODEL] (f32 or f16), W: [DMODEL, DMODEL] f32 (row n = output n).
// MODE 0: f16 out, head-reshaped [B,h,S,dk]
// MODE 1: f16 out, transposed    [B,h,dk,S]   (for V)
// MODE 2: f32 out, plain [MROWS, DMODEL]      (final)
enum { OUT_QK = 0, OUT_VT = 1, OUT_F32 = 2 };

template <typename TA, int MODE>
__global__ __launch_bounds__(256) void gemm_xwT(const TA* __restrict__ A,
                                                const float* __restrict__ W,
                                                const float* __restrict__ bias,
                                                void* __restrict__ outp) {
  const int lane = threadIdx.x & 31;
  const int wave = threadIdx.x >> 5;
  const int g  = lane >> 4;
  const int ml = lane & 15;
  const int m0 = blockIdx.x * 128 + wave * 16;   // 8 waves stacked in M
  const int n0 = blockIdx.y * 64;                // 4 N-tiles per wave

  const TA* arow = A + (size_t)(m0 + ml) * DMODEL;
  v8f acc[4] = {};

  for (int k0 = 0; k0 < DMODEL; k0 += 32) {
    __builtin_prefetch(arow + k0 + 128, 0, 3);   // global_prefetch_b8
    v16h af = a_frag(arow, k0, g);
#pragma unroll
    for (int t = 0; t < 4; ++t) {
      const float* wrow = W + (size_t)(n0 + t * 16 + ml) * DMODEL;
      acc[t] = wmma_f16(af, b_frag(wrow, k0 + 16 * g), acc[t]);
    }
  }

#pragma unroll
  for (int t = 0; t < 4; ++t) {
    const int n  = n0 + t * 16 + ml;
    const float bv = bias[n];
#pragma unroll
    for (int r = 0; r < 8; ++r) {
      const int   m = m0 + r + 8 * g;            // C-layout row
      const float v = acc[t][r] + bv;
      if (MODE == OUT_F32) {
        ((float*)outp)[(size_t)m * DMODEL + n] = v;
      } else {
        const int b = m >> 11, s = m & (SEQ - 1);
        const int h = n >> 6,  d = n & (DK - 1);
        if (MODE == OUT_QK)
          ((_Float16*)outp)[(((size_t)b * HEADS + h) * SEQ + s) * DK + d] =
              (_Float16)v;
        else
          ((_Float16*)outp)[(((size_t)b * HEADS + h) * DK + d) * SEQ + s] =
              (_Float16)v;
      }
    }
  }
}

// ---------------- flash attention: one wave per 16-query block --------------
__global__ __launch_bounds__(128) void attn_kernel(
    const _Float16* __restrict__ q, const _Float16* __restrict__ kmat,
    const _Float16* __restrict__ vT, const int* __restrict__ mask,
    _Float16* __restrict__ concat) {
  const int lane = threadIdx.x & 31;
  const int wave = threadIdx.x >> 5;
  const int g = lane >> 4, ml = lane & 15;
  const int wid = blockIdx.x * 4 + wave;
  const int qb = wid & (SEQ / 16 - 1);
  const int bh = wid >> 7;              // b*HEADS + h
  const int b  = bh >> 4;
  const int h  = bh & 15;
  const int s0 = qb * 16;

  __shared__ _Float16 lds[4 * 16 * 32];
  _Float16* pt = lds + wave * (16 * 32);     // per-wave 16x32 prob tile

  // Q A-fragments (dk=64 -> two K=32 steps), resident for whole loop
  const _Float16* qrow = q + ((size_t)bh * SEQ + s0 + ml) * DK;
  const v16h qf0 = a_frag(qrow, 0, g);
  const v16h qf1 = a_frag(qrow, 32, g);

  const _Float16* kbase = kmat + (size_t)bh * SEQ * DK;
  const _Float16* vbase = vT + (size_t)bh * DK * SEQ;
  const int*      mbase = mask + ((size_t)b * SEQ + s0) * SEQ;

  v8f o[4] = {};
  float mrun[8], lrun[8];
#pragma unroll
  for (int r = 0; r < 8; ++r) { mrun[r] = -3.0e38f; lrun[r] = 0.f; }

  for (int kc = 0; kc < SEQ; kc += 32) {
    // S = Q @ K^T for 32 keys (two 16x16 N-tiles, two K-steps each)
    v8f sc[2] = {};
#pragma unroll
    for (int nt = 0; nt < 2; ++nt) {
      const _Float16* krow = kbase + (size_t)(kc + nt * 16 + ml) * DK;
      sc[nt] = wmma_f16(qf0, b_frag(krow, 16 * g), sc[nt]);
      sc[nt] = wmma_f16(qf1, b_frag(krow, 32 + 16 * g), sc[nt]);
    }

    // scale, mask, online softmax update (rows live in 8 scalars/lane)
    float cmax[8];
#pragma unroll
    for (int r = 0; r < 8; ++r) {
      const int* mr = mbase + (r + 8 * g) * SEQ + kc + ml;
#pragma unroll
      for (int nt = 0; nt < 2; ++nt) {
        const float v = sc[nt][r] * 0.125f;          // 1/sqrt(dk)
        sc[nt][r] = (mr[nt * 16] != 0) ? v : -1e9f;
      }
      cmax[r] = fmaxf(sc[0][r], sc[1][r]);
#pragma unroll
      for (int d = 8; d >= 1; d >>= 1)               // max over 16-lane half
        cmax[r] = fmaxf(cmax[r], __shfl_xor(cmax[r], d, 32));
    }
#pragma unroll
    for (int r = 0; r < 8; ++r) {
      const float mnew  = fmaxf(mrun[r], cmax[r]);
      const float resc  = __expf(mrun[r] - mnew);
      const float p0 = __expf(sc[0][r] - mnew);
      const float p1 = __expf(sc[1][r] - mnew);
      sc[0][r] = p0; sc[1][r] = p1;
      float s = p0 + p1;
#pragma unroll
      for (int d = 8; d >= 1; d >>= 1) s += __shfl_xor(s, d, 32);
      lrun[r] = lrun[r] * resc + s;
      mrun[r] = mnew;
      o[0][r] *= resc; o[1][r] *= resc; o[2][r] *= resc; o[3][r] *= resc;
    }

    // C-layout probs -> LDS 16x32 f16 -> A-layout fragment
#pragma unroll
    for (int r = 0; r < 8; ++r) {
      pt[(r + 8 * g) * 32 + ml]      = (_Float16)sc[0][r];
      pt[(r + 8 * g) * 32 + 16 + ml] = (_Float16)sc[1][r];
    }
    asm volatile("s_wait_dscnt 0" ::: "memory");     // wave-wide DS RAW fence
    const v16h pf = a_frag((const _Float16*)(pt + ml * 32), 0, g);

    // O += P @ V   (V stored transposed [dk, S]: B-frags contiguous)
#pragma unroll
    for (int t = 0; t < 4; ++t) {
      const _Float16* vrow = vbase + (size_t)(t * 16 + ml) * SEQ + kc;
      o[t] = wmma_f16(pf, b_frag(vrow, 16 * g), o[t]);
    }
    asm volatile("s_wait_dscnt 0" ::: "memory");     // WAR vs next iter stores
  }

  // normalize and write concat [B,S,D] as f16 for the Wo GEMM
  _Float16* crow = concat + ((size_t)b * SEQ + s0) * DMODEL + h * DK;
#pragma unroll
  for (int r = 0; r < 8; ++r) {
    const float inv = 1.0f / lrun[r];
#pragma unroll
    for (int t = 0; t < 4; ++t)
      crow[(size_t)(r + 8 * g) * DMODEL + t * 16 + ml] =
          (_Float16)(o[t][r] * inv);
  }
}

// ---------------- host driver ----------------------------------------------
extern "C" void kernel_launch(void* const* d_in, const int* in_sizes, int n_in,
                              void* d_out, int out_size, void* d_ws,
                              size_t ws_size, hipStream_t stream) {
  (void)in_sizes; (void)n_in; (void)out_size; (void)ws_size;
  const float* query = (const float*)d_in[0];
  const float* key   = (const float*)d_in[1];
  const float* value = (const float*)d_in[2];
  const int*   mask  = (const int*)d_in[3];
  const float* Wq = (const float*)d_in[4];  const float* bq = (const float*)d_in[5];
  const float* Wk = (const float*)d_in[6];  const float* bk = (const float*)d_in[7];
  const float* Wv = (const float*)d_in[8];  const float* bv = (const float*)d_in[9];
  const float* Wo = (const float*)d_in[10]; const float* bo = (const float*)d_in[11];

  const size_t HS = (size_t)BATCH * HEADS * SEQ * DK;  // 8,388,608 elems
  _Float16* qh     = (_Float16*)d_ws;                  // [B,h,S,dk] f16
  _Float16* kh     = qh + HS;                          // [B,h,S,dk] f16
  _Float16* vTh    = kh + HS;                          // [B,h,dk,S] f16
  _Float16* concat = vTh + HS;                         // [B,S,D]    f16
  // total workspace: 4 * 16 MiB = 64 MiB

  dim3 gg(MROWS / 128, DMODEL / 64);
  gemm_xwT<float, OUT_QK><<<gg, 256, 0, stream>>>(query, Wq, bq, qh);
  gemm_xwT<float, OUT_QK><<<gg, 256, 0, stream>>>(key,   Wk, bk, kh);
  gemm_xwT<float, OUT_VT><<<gg, 256, 0, stream>>>(value, Wv, bv, vTh);

  attn_kernel<<<BATCH * HEADS * (SEQ / 16) / 4, 128, 0, stream>>>(
      qh, kh, vTh, mask, concat);

  gemm_xwT<_Float16, OUT_F32><<<gg, 256, 0, stream>>>(concat, Wo, bo,
                                                      (float*)d_out);
}
